// NutNet_81071802679474
// MI455X (gfx1250) — compile-verified, matrix operand order checked by reading, not run on hardware
//
// CDNA5 / gfx1250 implementation of the ST-QoS forward pass.
// All dense contractions go through one wave32 WMMA GEMM kernel using
// v_wmma_f32_16x16x32_f16. B matrices are pre-transposed+converted to f16
// (NxK64, zero-padded), staged per-block into LDS with double-buffered
// global_load_async_to_lds_b128 (+ s_wait_asynccnt), and read as LDS
// fragments. A streams from global as b128 loads with fp32->f16 cvt; only the
// A K-tail is masked (branchless clamped loads + selects). M/N edges handled
// by index clamping; garbage lanes are never stored. EXEC stays all-ones
// through every WMMA (no early returns in the GEMM).
//
// Param flattening assumption: top-level dict order for the 11 input arrays,
// then jax tree_leaves order for `params` (nested dicts sorted by key,
// lists/tuples in order) => 81 leaves at d_in[11..91].

#include <hip/hip_runtime.h>
#include <math.h>

typedef __attribute__((ext_vector_type(16))) _Float16 v16h;
typedef __attribute__((ext_vector_type(8)))  float    v8f;

#define TT 32      // time steps
#define NU 150     // users
#define MS 100     // servers (graph nodes)
#define KS 9       // slots
#define VS 2000    // services
#define DM 64      // model dim
#define NH 4       // heads
#define DH 16      // head dim
#define BB 4096    // batch
#define NE 800     // edges
#define KB 64      // GEMM K chunk staged in LDS

// Load one A fragment half-lane run: two contiguous 8-float chunks
// (p0 = K base, p1 = K base + 16), converted fp32 -> f16. 8-byte aligned by
// construction (even strides everywhere, k0 % 8 == 0).
static __device__ __forceinline__ v16h load_frag(const float* __restrict__ p0,
                                                 const float* __restrict__ p1)
{
  v16h f;
  const float2* q0 = (const float2*)p0;
  const float2* q1 = (const float2*)p1;
#pragma unroll
  for (int j = 0; j < 4; ++j) {
    float2 x = q0[j];
    f[2 * j]     = (_Float16)x.x;
    f[2 * j + 1] = (_Float16)x.y;
  }
#pragma unroll
  for (int j = 0; j < 4; ++j) {
    float2 x = q1[j];
    f[8 + 2 * j]     = (_Float16)x.x;
    f[8 + 2 * j + 1] = (_Float16)x.y;
  }
  return f;
}

// ---------------------------------------------------------------------------
// WMMA GEMM: C[M,N] = act( A[M,K] @ B[K,N] + bias (+ resid) ) (+= C optional)
// BT is B^T as f16, N x K64 (K64 = ceil(K/64)*64, zero padded). A rows go
// through a 3-level strided decomposition (D0,D1,D2)/(s0,s1,s2) + optional
// i0map gather (time gather, VALID-conv taps).
// Block = 256 threads = 8 waves: 8 consecutive 16-row M tiles sharing one
// 32-wide N group. B chunks (32 x KB f16 = 4KB) are staged into LDS with
// double-buffered global_load_async_to_lds_b128; each wave computes a 16x32
// tile (two accumulators) per 32-K step. flags: 1=relu, 2=accum, 4=resid.
// ---------------------------------------------------------------------------
__global__ __launch_bounds__(256) void wmma_gemm(
    const float* __restrict__ A, const int* __restrict__ i0map,
    int D0, int D1, int D2, long s0, long s1, long s2,
    const _Float16* __restrict__ BT,
    const float* __restrict__ bias,
    const float* __restrict__ resid, int ldr,
    float* __restrict__ C, int ldc,
    int M, int N, int K, int K64, int flags)
{
  __shared__ _Float16 sB[2][32 * KB];                 // 8 KB double buffer
  const int tid  = threadIdx.x;
  const int lane = tid & 31, wave = tid >> 5;
  const int tilesN = (N + 31) >> 5;
  const int tilesM = (M + 15) >> 4;
  const int nGrp = blockIdx.x % tilesN;
  const int mBlk = blockIdx.x / tilesN;
  int tM = mBlk * 8 + wave;
  const bool active = tM < tilesM;                    // inactive waves still barrier
  if (!active) tM = tilesM - 1;
  const int tN0 = nGrp << 5;
  const int hi = lane >> 4, lm = lane & 15;

  // A row, clamped: OOB rows compute garbage that is never stored.
  int rowA = tM * 16 + lm; if (rowA >= M) rowA = M - 1;
  long offA;
  {
    const int d12 = D1 * D2;
    int i0 = rowA / d12; int rem = rowA - i0 * d12;
    int i1 = rem / D2;   int i2 = rem - i1 * D2;
    if (i0map) i0 = i0map[i0];
    offA = (long)i0 * s0 + (long)i1 * s1 + (long)i2 * s2;
  }
  const float* __restrict__ Ap = A + offA;

  // Async B loader: each thread stages one 16B run (8 f16) per chunk.
  const int lcol = tid >> 3;                           // 0..31 local column
  const int lk   = (tid & 7) * 8;                      // 0,8,...,56 K offset
  int gcol = tN0 + lcol; if (gcol >= N) gcol = N - 1;  // clamped (dup loads ok)
  const _Float16* gB = BT + (long)gcol * K64 + lk;
  const unsigned lds0 = (unsigned)(uintptr_t)&sB[0][lcol * KB + lk]; // ISA 10.2:
  const unsigned lds1 = (unsigned)(uintptr_t)&sB[1][lcol * KB + lk]; // LDS = addr[31:0]

  v8f acc0 = {0.f, 0.f, 0.f, 0.f, 0.f, 0.f, 0.f, 0.f};
  v8f acc1 = {0.f, 0.f, 0.f, 0.f, 0.f, 0.f, 0.f, 0.f};
  const int b0base = lm * KB;                          // B fragment cols in LDS
  const int b1base = (lm + 16) * KB;
  const int nChunks = K64 / KB;

  {
    unsigned long long g = (unsigned long long)(uintptr_t)gB;
    asm volatile("global_load_async_to_lds_b128 %0, %1, off"
                 :: "v"(lds0), "v"(g) : "memory");
  }
  for (int c = 0; c < nChunks; ++c) {
    const int cur = c & 1;
    if (c + 1 < nChunks) {
      unsigned long long g = (unsigned long long)(uintptr_t)(gB + (long)(c + 1) * KB);
      asm volatile("global_load_async_to_lds_b128 %0, %1, off"
                   :: "v"(cur ? lds0 : lds1), "v"(g) : "memory");
      __builtin_prefetch(Ap + (c + 1) * KB, 0, 1);     // A stream hint
      asm volatile("s_wait_asynccnt 0x1" ::: "memory"); // chunk c arrived
    } else {
      asm volatile("s_wait_asynccnt 0x0" ::: "memory");
    }
    __syncthreads();                                   // block sees sB[cur]
#pragma unroll
    for (int s = 0; s < 2; ++s) {
      const int kb = c * KB + s * 32;
      const int kl = s * 32 + hi * 8;
      v16h a, b0, b1;
#pragma unroll
      for (int j = 0; j < 8; ++j) {                    // LDS fragment reads
        b0[j]     = sB[cur][b0base + kl + j];
        b0[j + 8] = sB[cur][b0base + kl + 16 + j];
        b1[j]     = sB[cur][b1base + kl + j];
        b1[j + 8] = sB[cur][b1base + kl + 16 + j];
      }
      if (kb + 32 <= K) {                              // uniform branch
        a = load_frag(Ap + kb + hi * 8, Ap + kb + hi * 8 + 16);
      } else {                                         // masked A tail (B is 0-padded)
        const _Float16 z = (_Float16)0.0f;
#pragma unroll
        for (int j = 0; j < 8; ++j) {
          const int ka = kb + hi * 8 + j, kc = ka + 16;
          const int kaC = ka < K ? ka : 0, kcC = kc < K ? kc : 0;
          float va = Ap[kaC], vb = Ap[kcC];
          a[j]     = ka < K ? (_Float16)va : z;
          a[j + 8] = kc < K ? (_Float16)vb : z;
        }
      }
      acc0 = __builtin_amdgcn_wmma_f32_16x16x32_f16(false, a, false, b0, (short)0, acc0, false, false);
      acc1 = __builtin_amdgcn_wmma_f32_16x16x32_f16(false, a, false, b1, (short)0, acc1, false, false);
    }
    __syncthreads();                                   // done reading sB[cur]
  }

#pragma unroll
  for (int g = 0; g < 2; ++g) {
    const int col = tN0 + g * 16 + lm;
    const bool cOK = active && col < N;
    float bs = 0.f;
    if (bias && cOK) bs = bias[col];
#pragma unroll
    for (int r8 = 0; r8 < 8; ++r8) {
      const int rowC = tM * 16 + r8 + hi * 8;   // C/D layout: lanes16-31 -> M=r+8
      if (rowC < M && cOK) {
        float v = (g ? acc1[r8] : acc0[r8]) + bs;
        const long ci = (long)rowC * ldc + col;
        if (flags & 4) v += resid[(long)rowC * ldr + col];
        if (flags & 1) v = fmaxf(v, 0.f);
        if (flags & 2) v += C[ci];
        C[ci] = v;
      }
    }
  }
}

// B (KxN fp32) -> BT (NxK64 f16, zero padded)
__global__ void transpose_pad_f16(const float* __restrict__ B, _Float16* __restrict__ BT,
                                  int K, int N, int K64)
{
  long i = (long)blockIdx.x * blockDim.x + threadIdx.x;
  if (i >= (long)N * K64) return;
  int n = (int)(i / K64), k = (int)(i % K64);
  BT[i] = (k < K) ? (_Float16)B[(long)k * N + n] : (_Float16)0.0f;
}

// ---------------------------------------------------------------------------
// Tables: DFT matrices (64-pt, 9-pt) and positional encoding (9 x 64)
// ---------------------------------------------------------------------------
__global__ void init_tables(float* Cd, float* Sd, float* Ck, float* Sk, float* pe)
{
  const float PI2 = 6.28318530717958647692f;
  int i = blockIdx.x * blockDim.x + threadIdx.x;
  if (i < 64 * 64) { int p = i >> 6, q = i & 63; float a = PI2 * (float)(p * q) / 64.f;
                     Cd[i] = cosf(a); Sd[i] = sinf(a); }
  if (i < 81)      { int p = i / 9, q = i % 9; float a = PI2 * (float)(p * q) / 9.f;
                     Ck[i] = cosf(a); Sk[i] = sinf(a); }
  if (i < KS * 64) { int pos = i >> 6, q = i & 63;
                     float a = (float)pos / powf(10000.f, 2.f * floorf(q * 0.5f) / 64.f);
                     pe[i] = (q & 1) ? cosf(a) : sinf(a); }
}

// ---------------------------------------------------------------------------
// unique(info[:,3]) emulation: presence flags -> ranks -> gather map
// ---------------------------------------------------------------------------
__global__ void uniq_flags(const int* __restrict__ info, int* flags)
{ int i = blockIdx.x * blockDim.x + threadIdx.x; if (i < BB) flags[info[i * 4 + 3] - KS] = 1; }

__global__ void uniq_scan(const int* flags, int* rank, int* uqidx)
{
  if (threadIdx.x || blockIdx.x) return;
  int cnt = 0;
  for (int t = 0; t < TT; ++t) { if (flags[t]) { rank[t] = cnt; uqidx[cnt] = t; ++cnt; } else rank[t] = 0; }
  for (int j = cnt; j < TT; ++j) uqidx[j] = 0;     // fill_value=k -> idx 0
}

__global__ void uniq_inv(const int* __restrict__ info, const int* rank, int* inv)
{ int i = blockIdx.x * blockDim.x + threadIdx.x; if (i < BB) inv[i] = rank[info[i * 4 + 3] - KS]; }

// ---------------------------------------------------------------------------
// Embedding tables
// ---------------------------------------------------------------------------
__global__ void build_svc(const int* __restrict__ svc_attr,
                          const float* t0, const float* t1, const float* t2, const float* t3,
                          float* out)
{
  int i = blockIdx.x * blockDim.x + threadIdx.x;
  if (i >= VS * 64) return;
  int r = i >> 6, c = i & 63, tb = c >> 4, cc = c & 15;
  int a = svc_attr[r * 4 + tb] - (tb > 0 ? 1 : 0);
  const float* tab = tb == 0 ? t0 : tb == 1 ? t1 : tb == 2 ? t2 : t3;
  out[i] = tab[a * 16 + cc];
}

__global__ void build_srv(const float* __restrict__ srv_attr,
                          const float* t0, const float* t1, const float* t2, const float* t3,
                          float* out)
{
  int i = blockIdx.x * blockDim.x + threadIdx.x;
  if (i >= MS * 67) return;
  int r = i / 67, c = i % 67;
  if (c < 64) {
    int tb = c >> 4, cc = c & 15;
    int a = (int)srv_attr[r * 7 + tb] - (tb > 0 ? 1 : 0);
    const float* tab = tb == 0 ? t0 : tb == 1 ? t1 : tb == 2 ? t2 : t3;
    out[i] = tab[a * 16 + cc];
  } else out[i] = srv_attr[r * 7 + 4 + (c - 64)];
}

// concat [usr_emb | tra | u_inv@svc_emb] -> (43200, 84)
__global__ void build_ubig(const float* __restrict__ usr_tab, const int* __restrict__ usr_attr,
                           const float* __restrict__ tra, const float* __restrict__ tmpU,
                           const int* __restrict__ uqidx, float* out)
{
  long i = (long)blockIdx.x * blockDim.x + threadIdx.x;
  if (i >= (long)TT * NU * KS * 84) return;
  long r = i / 84; int c = (int)(i % 84);
  int t = (int)(r / (NU * KS)); int rem = (int)(r % (NU * KS));
  int ni = rem / KS, ki = rem % KS;
  float v;
  if (c < 16)      v = usr_tab[usr_attr[ni] * 16 + c];
  else if (c < 20) v = tra[(((long)uqidx[t] * NU + ni) * KS + ki) * 4 + (c - 16)];
  else             v = tmpU[r * 64 + (c - 20)];
  out[i] = v;
}

// concat [srv_emb | srv | e_inv@svc_emb] -> (28800, 134)
__global__ void build_sbig(const float* __restrict__ srvEmb, const float* __restrict__ srv,
                           const float* __restrict__ tmpS, const int* __restrict__ uqidx, float* out)
{
  long i = (long)blockIdx.x * blockDim.x + threadIdx.x;
  if (i >= (long)TT * MS * KS * 134) return;
  long r = i / 134; int c = (int)(i % 134);
  int t = (int)(r / (MS * KS)); int rem = (int)(r % (MS * KS));
  int mi = rem / KS, ki = rem % KS;
  float v;
  if (c < 67)      v = srvEmb[mi * 67 + c];
  else if (c < 70) v = srv[(((long)uqidx[t] * MS + mi) * KS + ki) * 3 + (c - 67)];
  else             v = tmpS[r * 64 + (c - 70)];
  out[i] = v;
}

__global__ void add_pe(float* X, const float* __restrict__ pe, long M)
{
  long i = (long)blockIdx.x * blockDim.x + threadIdx.x;
  if (i >= M * 64) return;
  int q = (int)(i & 63); long r = i >> 6;
  X[i] += pe[(int)(r % KS) * 64 + q];
}

// ---------------------------------------------------------------------------
// FNet mixer: out = x + real(DFT2(x)) over each 9x64 slab.
// real(F_k X F_d) = Ck X Cd - Sk X Sd  (F = C - iS)
// ---------------------------------------------------------------------------
__global__ void fnet_fft_add(const float* __restrict__ X,
                             const float* __restrict__ Cd, const float* __restrict__ Sd,
                             const float* __restrict__ Ck, const float* __restrict__ Sk,
                             float* __restrict__ Out)
{
  __shared__ float sx[KS * 64], sy[KS * 64], sz[KS * 64];
  const float* x = X + (long)blockIdx.x * (KS * 64);
  for (int i = threadIdx.x; i < KS * 64; i += blockDim.x) sx[i] = x[i];
  __syncthreads();
  int q = threadIdx.x;                 // 64 threads, one output column each
  for (int a = 0; a < KS; ++a) {
    float ys = 0.f, zs = 0.f;
    for (int b = 0; b < 64; ++b) {
      float xv = sx[a * 64 + b];
      ys += xv * Cd[q * 64 + b];
      zs += xv * Sd[q * 64 + b];
    }
    sy[a * 64 + q] = ys; sz[a * 64 + q] = zs;
  }
  __syncthreads();
  float* o = Out + (long)blockIdx.x * (KS * 64);
  for (int p = 0; p < KS; ++p) {
    float h = 0.f;
    for (int a = 0; a < KS; ++a)
      h += Ck[p * 9 + a] * sy[a * 64 + q] - Sk[p * 9 + a] * sz[a * 64 + q];
    o[p * 64 + q] = sx[p * 64 + q] + h;
  }
}

// ---------------------------------------------------------------------------
// LayerNorm over last dim 64; one wave32 per row, shfl_xor tree reduction.
// ---------------------------------------------------------------------------
__global__ void layernorm64(const float* __restrict__ X, const float* __restrict__ g,
                            const float* __restrict__ c, float* __restrict__ Y, long M)
{
  long wid = ((long)blockIdx.x * blockDim.x + threadIdx.x) >> 5;
  int lane = threadIdx.x & 31;
  if (wid >= M) return;
  const float* x = X + wid * 64;
  float a0 = x[lane], a1 = x[lane + 32];
  float s = a0 + a1;
#pragma unroll
  for (int o = 16; o; o >>= 1) s += __shfl_xor(s, o, 32);
  float mu = s * (1.f / 64.f);
  float d0 = a0 - mu, d1 = a1 - mu;
  float vs = d0 * d0 + d1 * d1;
#pragma unroll
  for (int o = 16; o; o >>= 1) vs += __shfl_xor(vs, o, 32);
  float inv = rsqrtf(vs * (1.f / 64.f) + 1e-5f);
  float* y = Y + wid * 64;
  y[lane]      = d0 * inv * g[lane]      + c[lane];
  y[lane + 32] = d1 * inv * g[lane + 32] + c[lane + 32];
}

// ---------------------------------------------------------------------------
// Multi-head attention with online softmax. One block per (t, slot, head);
// one thread per query row m. Layouts stay (T, rows, k, 64); heads are
// 16-column stripes. mask gathered through uqidx.
// ---------------------------------------------------------------------------
__global__ void attention(const float* __restrict__ Q, const float* __restrict__ Km,
                          const float* __restrict__ V, const float* __restrict__ mask,
                          const int* __restrict__ uqidx, float* __restrict__ O)
{
  int blk = blockIdx.x;
  int h = blk % NH; int tk = blk / NH;
  int t = tk / KS, ksl = tk % KS;
  int mq = threadIdx.x;
  if (mq >= MS) return;
  const long qrow = ((long)t * MS + mq) * KS + ksl;
  const float* q = Q + qrow * 64 + h * DH;
  float qr[DH];
#pragma unroll
  for (int e = 0; e < DH; ++e) qr[e] = q[e] * 0.25f;     // 1/sqrt(dh)
  const float* mrow = mask + (((long)uqidx[t] * KS + ksl) * MS + mq) * NU;
  float mx = -1e30f, ssum = 0.f, acc[DH];
#pragma unroll
  for (int e = 0; e < DH; ++e) acc[e] = 0.f;
  for (int nk = 0; nk < NU; ++nk) {
    const long krow = ((long)t * NU + nk) * KS + ksl;
    const float* kp = Km + krow * 64 + h * DH;
    float s = 0.f;
#pragma unroll
    for (int e = 0; e < DH; ++e) s += qr[e] * kp[e];
    s += (1.f - mrow[nk]) * -1e9f;
    float nmx = fmaxf(mx, s);
    float sc = expf(mx - nmx);
    float w  = expf(s - nmx);
    ssum = ssum * sc + w;
    const float* vp = V + krow * 64 + h * DH;
#pragma unroll
    for (int e = 0; e < DH; ++e) acc[e] = acc[e] * sc + w * vp[e];
    mx = nmx;
  }
  float inv = 1.f / ssum;
  float* o = O + qrow * 64 + h * DH;
#pragma unroll
  for (int e = 0; e < DH; ++e) o[e] = acc[e] * inv;
}

// tem (T,9,m,128) = concat[ srvOut (T,m,9,64) , decOut (T,m,9,64) ] transposed
__global__ void build_tem(const float* __restrict__ srvF, const float* __restrict__ decF,
                          float* __restrict__ tem)
{
  long i = (long)blockIdx.x * blockDim.x + threadIdx.x;
  if (i >= (long)TT * KS * MS * 128) return;
  int c = (int)(i & 127);
  int w = (int)((i >> 7) % MS);
  int h = (int)((i / (128 * MS)) % KS);
  int t = (int)(i / ((long)128 * MS * KS));
  long src = (((long)t * MS + w) * KS + h) * 64;
  tem[i] = (c < 64) ? srvF[src + c] : decF[src + c - 64];
}

// GLU-ish combine for _tconv: out = relu(P*sigmoid(Q)+R) * scale
__global__ void glu_act(const float* __restrict__ P, const float* __restrict__ Q,
                        const float* __restrict__ R, float* __restrict__ O,
                        long n, float scale)
{
  long i = (long)blockIdx.x * blockDim.x + threadIdx.x;
  if (i >= n) return;
  float s = 1.f / (1.f + expf(-Q[i]));
  O[i] = fmaxf(P[i] * s + R[i], 0.f) * scale;
}

// graph degree / normalization / edge weights
__global__ void deg_kernel(const int* __restrict__ src, float* deg)
{ int i = blockIdx.x * blockDim.x + threadIdx.x; if (i < NE) atomicAdd(&deg[src[i]], 1.f); }

__global__ void dis_kernel(float* deg)
{ int i = threadIdx.x; if (i < MS) deg[i] = deg[i] > 0.f ? rsqrtf(fmaxf(deg[i], 1.f)) : 0.f; }

__global__ void w_kernel(const int* __restrict__ src, const int* __restrict__ dst,
                         const float* __restrict__ dis, float* w)
{ int i = blockIdx.x * blockDim.x + threadIdx.x; if (i < NE) w[i] = -dis[src[i]] * dis[dst[i]]; }

// prop(Z): out[dst] += w_e * Z[src], node features of length S
__global__ void cheb_prop(const int* __restrict__ src, const int* __restrict__ dst,
                          const float* __restrict__ w, const float* __restrict__ Z,
                          float* __restrict__ O, int S)
{
  long i = (long)blockIdx.x * blockDim.x + threadIdx.x;
  if (i >= (long)NE * S) return;
  int e = (int)(i / S), f = (int)(i % S);
  atomicAdd(&O[(long)dst[e] * S + f], w[e] * Z[(long)src[e] * S + f]);
}

__global__ void z2_combine(const float* __restrict__ P2, const float* __restrict__ Z0,
                           float* __restrict__ Z2, long n)
{ long i = (long)blockIdx.x * blockDim.x + threadIdx.x; if (i < n) Z2[i] = 2.f * P2[i] - Z0[i]; }

// (T,H,m,C) -> (m,T,H,C)
__global__ void to_nodes(const float* __restrict__ X, float* __restrict__ Z,
                         int Td, int H, int Md, int C)
{
  long i = (long)blockIdx.x * blockDim.x + threadIdx.x;
  long tot = (long)Td * H * Md * C;
  if (i >= tot) return;
  int c = (int)(i % C); long r = i / C;
  int w = (int)(r % Md); r /= Md;
  int h = (int)(r % H);  int t = (int)(r / H);
  Z[((((long)w * Td + t) * H + h) * C) + c] = X[i];
}

// (m,T,H,C) -> (T,H,m,C), with relu
__global__ void from_nodes_relu(const float* __restrict__ Z, float* __restrict__ X,
                                int Td, int H, int Md, int C)
{
  long i = (long)blockIdx.x * blockDim.x + threadIdx.x;
  long tot = (long)Td * H * Md * C;
  if (i >= tot) return;
  int c = (int)(i % C); long r = i / C;
  int w = (int)(r % Md); r /= Md;
  int h = (int)(r % H);  int t = (int)(r / H);
  X[i] = fmaxf(Z[((((long)w * Td + t) * H + h) * C) + c], 0.f);
}

// final feature concat: [u_sel | s_sel | c_sel] -> (b, 192)
__global__ void build_final(const float* __restrict__ usrF, const float* __restrict__ tem2,
                            const float* __restrict__ svcEmb, const int* __restrict__ info,
                            const int* __restrict__ inv, float* __restrict__ X)
{
  long i = (long)blockIdx.x * blockDim.x + threadIdx.x;
  if (i >= (long)BB * 192) return;
  int rb = (int)(i / 192), c = (int)(i % 192);
  int u = info[rb * 4 + 0], s = info[rb * 4 + 1], cs = info[rb * 4 + 2];
  int t = inv[rb];
  float v;
  if (c < 64)       v = usrF[((((long)t * NU + u) * KS) + (KS - 1)) * 64 + c];
  else if (c < 128) v = tem2[((long)t * MS + s) * 64 + (c - 64)];
  else              v = svcEmb[(long)cs * 64 + (c - 128)];
  X[i] = v;
}

// ===========================================================================
extern "C" void kernel_launch(void* const* d_in, const int* in_sizes, int n_in,
                              void* d_out, int out_size, void* d_ws, size_t ws_size,
                              hipStream_t stream)
{
  (void)in_sizes; (void)n_in; (void)out_size; (void)ws_size;
  // ----- inputs -----
  const float* tra      = (const float*)d_in[0];
  const float* u_inv    = (const float*)d_in[1];
  const float* srv      = (const float*)d_in[2];
  const float* e_inv    = (const float*)d_in[3];
  const float* mask     = (const float*)d_in[4];
  const int*   info     = (const int*)  d_in[5];
  const int*   edge     = (const int*)  d_in[7];   // (2,800): src then dst
  const int*   usr_attr = (const int*)  d_in[8];
  const float* srv_attr = (const float*)d_in[9];
  const int*   svc_attr = (const int*)  d_in[10];
  // ----- params (tree_leaves order; see header comment) -----
  auto pf = [&](int i) { return (const float*)d_in[i]; };
  const float *dec_W1=pf(11), *dec_W2=pf(12), *dec_Wk=pf(13), *dec_Wo=pf(14),
              *dec_Wq=pf(15), *dec_Wv=pf(16), *dec_b1=pf(17), *dec_b2=pf(18),
              *dec_bk=pf(19), *dec_bo=pf(20), *dec_bq=pf(21), *dec_bv=pf(22),
              *dec_c1=pf(23), *dec_c2=pf(24), *dec_g1=pf(25), *dec_g2=pf(26);
  const float *qW0=pf(27), *qb0=pf(28), *qW1=pf(29), *qb1=pf(30),
              *qW2=pf(31), *qb2=pf(32), *qW3=pf(33), *qb3=pf(34);
  const float *sfn_W1=pf(35), *sfn_W2=pf(36), *sfn_b1=pf(37), *sfn_b2=pf(38),
              *sfn_c1=pf(39), *sfn_c2=pf(40), *sfn_g1=pf(41), *sfn_g2=pf(42);
  const float *sp_W=pf(43), *sp_b=pf(44);
  const float *srvT0=pf(45), *srvT1=pf(46), *srvT2=pf(47), *srvT3=pf(48);
  const float *st1_cW=pf(49), *st1_cb=pf(50);
  const float *st1_t1W[3]={pf(51),pf(53),pf(55)}; const float *st1_t1b[3]={pf(52),pf(54),pf(56)};
  const float *st1_t2W[3]={pf(57),pf(59),pf(61)}; const float *st1_t2b[3]={pf(58),pf(60),pf(62)};
  const float *st2_cW=pf(63), *st2_cb=pf(64);
  const float *st2_t1W[3]={pf(65),pf(67),pf(69)}; const float *st2_t1b[3]={pf(66),pf(68),pf(70)};
  const float *st2_t2W[3]={pf(71),pf(73),pf(75)}; const float *st2_t2b[3]={pf(72),pf(74),pf(76)};
  const float *svcT0=pf(77), *svcT1=pf(78), *svcT2=pf(79), *svcT3=pf(80);
  const float *ufn_W1=pf(81), *ufn_W2=pf(82), *ufn_b1=pf(83), *ufn_b2=pf(84),
              *ufn_c1=pf(85), *ufn_c2=pf(86), *ufn_g1=pf(87), *ufn_g2=pf(88);
  const float *up_W=pf(89), *up_b=pf(90), *usr_tab=pf(91);

  // ----- workspace bump allocator -----
  uintptr_t wsp = (uintptr_t)d_ws;
  auto allocf = [&](size_t n) { wsp = (wsp + 255) & ~(uintptr_t)255; float* p = (float*)wsp; wsp += n * 4; return p; };
  auto alloci = [&](size_t n) { wsp = (wsp + 255) & ~(uintptr_t)255; int*   p = (int*)wsp;   wsp += n * 4; return p; };
  auto alloch = [&](size_t n) { wsp = (wsp + 255) & ~(uintptr_t)255; _Float16* p = (_Float16*)wsp; wsp += n * 2; return p; };

  const long MU = (long)TT * NU * KS;   // 43200
  const long MSr = (long)TT * MS * KS;  // 28800
  float *Cd = allocf(4096), *Sd = allocf(4096), *Ck = allocf(81), *Sk = allocf(81), *pe = allocf(KS * 64);
  int *flags = alloci(32), *rank = alloci(32), *uqidx = alloci(32), *inv = alloci(BB);
  float *svcEmb = allocf((size_t)VS * 64), *srvEmb = allocf(MS * 67);
  float *tmpU = allocf(MU * 64), *Ubig = allocf(MU * 84);
  float *usrX = allocf(MU * 64), *usrH = allocf(MU * 64), *usrLN = allocf(MU * 64),
        *usrZ1 = allocf(MU * 128), *usrFFN = allocf(MU * 64), *usrOut = allocf(MU * 64);
  float *tmpS = allocf(MSr * 64), *Sbig = allocf(MSr * 134);
  float *srvX = allocf(MSr * 64), *srvH = allocf(MSr * 64), *srvLN = allocf(MSr * 64),
        *srvZ1 = allocf(MSr * 128), *srvFFN = allocf(MSr * 64), *srvOut = allocf(MSr * 64);
  float *qP = allocf(MSr * 64), *kP = allocf(MU * 64), *vP = allocf(MU * 64), *attO = allocf(MSr * 64);
  float *oP = allocf(MSr * 64), *decLN = allocf(MSr * 64), *decZ1 = allocf(MSr * 128),
        *decFFN = allocf(MSr * 64), *decOut = allocf(MSr * 64);
  float *tem = allocf((size_t)TT * KS * MS * 128);
  float *deg = allocf(MS), *wE = allocf(NE);
  const int MC1 = TT * 7 * MS;   // 22400  st1 t1 / cheb rows
  const int MD1 = TT * 5 * MS;   // 16000  st1 t2 rows
  const int ME1 = TT * 3 * MS;   // 9600   st2 t1 / cheb rows
  const int MF1 = TT * 1 * MS;   // 3200   st2 t2 rows
  float *cP = allocf((size_t)MC1*128), *cQ = allocf((size_t)MC1*128), *cR = allocf((size_t)MC1*128), *cX = allocf((size_t)MC1*128);
  float *Za = allocf((size_t)MC1*128), *Zb = allocf((size_t)MC1*128), *Zp = allocf((size_t)MC1*128),
        *Zc = allocf((size_t)MC1*128), *OutC = allocf((size_t)MC1*128), *Xr = allocf((size_t)MC1*128);
  float *dP = allocf((size_t)MD1*64), *dQ = allocf((size_t)MD1*64), *dR = allocf((size_t)MD1*64), *Y1 = allocf((size_t)MD1*64);
  float *eP = allocf((size_t)ME1*128), *eQ = allocf((size_t)ME1*128), *eR = allocf((size_t)ME1*128), *eX = allocf((size_t)ME1*128);
  float *Za2 = allocf((size_t)ME1*128), *Zb2 = allocf((size_t)ME1*128), *Zp2 = allocf((size_t)ME1*128),
        *Zc2 = allocf((size_t)ME1*128), *OutC2 = allocf((size_t)ME1*128), *Xr2 = allocf((size_t)ME1*128);
  float *fP = allocf((size_t)MF1*64), *fQ = allocf((size_t)MF1*64), *fR = allocf((size_t)MF1*64), *tem2 = allocf((size_t)MF1*64);
  float *xF = allocf((size_t)BB*192), *m1 = allocf((size_t)BB*32), *m2 = allocf((size_t)BB*16), *m3 = allocf((size_t)BB*8);

  auto eg = [](long n, int bs) { return dim3((unsigned)((n + bs - 1) / bs)); };
  // pre-transpose+convert a KxN weight to f16 NxK64 (zero padded), return BT
  auto mkT = [&](const float* B, int K, int N) {
    int K64 = ((K + 63) / 64) * 64;
    _Float16* BT = alloch((size_t)N * K64);
    transpose_pad_f16<<<eg((long)N * K64, 256), 256, 0, stream>>>(B, BT, K, N, K64);
    return (const _Float16*)BT;
  };
  // GEMM launcher: 8 M-tiles x one 32-wide N group per block (256 threads)
  auto gemm3 = [&](const float* A, const int* i0map, int D0, int D1, int D2,
                   long s0, long s1, long s2, const _Float16* BT,
                   const float* bias, const float* resid, float* C, int ldc,
                   int M, int N, int K, int fl) {
    int tilesM = (M + 15) / 16, tilesN = (N + 31) / 32;
    int K64 = ((K + 63) / 64) * 64;
    wmma_gemm<<<dim3((unsigned)(((tilesM + 7) / 8) * tilesN)), dim3(256), 0, stream>>>(
        A, i0map, D0, D1, D2, s0, s1, s2, BT, bias, resid, ldc, C, ldc, M, N, K, K64, fl);
  };
  auto gemm = [&](const float* A, int lda, const _Float16* BT, const float* bias,
                  const float* resid, float* C, int ldc, int M, int N, int K, int fl) {
    gemm3(A, nullptr, M, 1, 1, lda, 0, 0, BT, bias, resid, C, ldc, M, N, K, fl);
  };

  // ---- tables, unique(info[:,3]) ----
  init_tables<<<eg(4096, 256), 256, 0, stream>>>(Cd, Sd, Ck, Sk, pe);
  hipMemsetAsync(flags, 0, 32 * sizeof(int), stream);
  uniq_flags<<<eg(BB, 256), 256, 0, stream>>>(info, flags);
  uniq_scan<<<1, 1, 0, stream>>>(flags, rank, uqidx);
  uniq_inv<<<eg(BB, 256), 256, 0, stream>>>(info, rank, inv);

  // ---- embeddings ----
  build_svc<<<eg((long)VS * 64, 256), 256, 0, stream>>>(svc_attr, svcT0, svcT1, svcT2, svcT3, svcEmb);
  build_srv<<<eg((long)MS * 67, 256), 256, 0, stream>>>(srv_attr, srvT0, srvT1, srvT2, srvT3, srvEmb);
  const _Float16* svcEmbT = mkT(svcEmb, VS, 64);

  // ---- user branch: big embedding GEMM -> projection -> +pe -> FNet ----
  gemm3(u_inv, uqidx, TT, NU * KS, 1, (long)NU * KS * VS, VS, 0,
        svcEmbT, nullptr, nullptr, tmpU, 64, (int)MU, 64, VS, 0);
  build_ubig<<<eg(MU * 84, 256), 256, 0, stream>>>(usr_tab, usr_attr, tra, tmpU, uqidx, Ubig);
  gemm(Ubig, 84, mkT(up_W, 84, 64), up_b, nullptr, usrX, 64, (int)MU, 64, 84, 0);
  add_pe<<<eg(MU * 64, 256), 256, 0, stream>>>(usrX, pe, MU);
  fnet_fft_add<<<dim3(TT * NU), dim3(64), 0, stream>>>(usrX, Cd, Sd, Ck, Sk, usrH);
  layernorm64<<<eg(MU * 32, 256), 256, 0, stream>>>(usrH, ufn_g1, ufn_c1, usrLN, MU);
  gemm(usrLN, 64, mkT(ufn_W1, 64, 128), ufn_b1, nullptr, usrZ1, 128, (int)MU, 128, 64, 1);
  gemm(usrZ1, 128, mkT(ufn_W2, 128, 64), ufn_b2, usrLN, usrFFN, 64, (int)MU, 64, 128, 4);
  layernorm64<<<eg(MU * 32, 256), 256, 0, stream>>>(usrFFN, ufn_g2, ufn_c2, usrOut, MU);

  // ---- server branch ----
  gemm3(e_inv, uqidx, TT, MS * KS, 1, (long)MS * KS * VS, VS, 0,
        svcEmbT, nullptr, nullptr, tmpS, 64, (int)MSr, 64, VS, 0);
  build_sbig<<<eg(MSr * 134, 256), 256, 0, stream>>>(srvEmb, srv, tmpS, uqidx, Sbig);
  gemm(Sbig, 134, mkT(sp_W, 134, 64), sp_b, nullptr, srvX, 64, (int)MSr, 64, 134, 0);
  add_pe<<<eg(MSr * 64, 256), 256, 0, stream>>>(srvX, pe, MSr);
  fnet_fft_add<<<dim3(TT * MS), dim3(64), 0, stream>>>(srvX, Cd, Sd, Ck, Sk, srvH);
  layernorm64<<<eg(MSr * 32, 256), 256, 0, stream>>>(srvH, sfn_g1, sfn_c1, srvLN, MSr);
  gemm(srvLN, 64, mkT(sfn_W1, 64, 128), sfn_b1, nullptr, srvZ1, 128, (int)MSr, 128, 64, 1);
  gemm(srvZ1, 128, mkT(sfn_W2, 128, 64), sfn_b2, srvLN, srvFFN, 64, (int)MSr, 64, 128, 4);
  layernorm64<<<eg(MSr * 32, 256), 256, 0, stream>>>(srvFFN, sfn_g2, sfn_c2, srvOut, MSr);

  // ---- decoder (cross attention srv->usr) ----
  gemm(srvOut, 64, mkT(dec_Wq, 64, 64), dec_bq, nullptr, qP, 64, (int)MSr, 64, 64, 0);
  gemm(usrOut, 64, mkT(dec_Wk, 64, 64), dec_bk, nullptr, kP, 64, (int)MU, 64, 64, 0);
  gemm(usrOut, 64, mkT(dec_Wv, 64, 64), dec_bv, nullptr, vP, 64, (int)MU, 64, 64, 0);
  attention<<<dim3(TT * KS * NH), dim3(128), 0, stream>>>(qP, kP, vP, mask, uqidx, attO);
  gemm(attO, 64, mkT(dec_Wo, 64, 64), dec_bo, srvOut, oP, 64, (int)MSr, 64, 64, 4);
  layernorm64<<<eg(MSr * 32, 256), 256, 0, stream>>>(oP, dec_g1, dec_c1, decLN, MSr);
  gemm(decLN, 64, mkT(dec_W1, 64, 128), dec_b1, nullptr, decZ1, 128, (int)MSr, 128, 64, 1);
  gemm(decZ1, 128, mkT(dec_W2, 128, 64), dec_b2, decLN, decFFN, 64, (int)MSr, 64, 128, 4);
  layernorm64<<<eg(MSr * 32, 256), 256, 0, stream>>>(decFFN, dec_g2, dec_c2, decOut, MSr);

  // ---- tem = concat(srv, dec) in (T,9,m,128) ----
  build_tem<<<eg((long)TT * KS * MS * 128, 256), 256, 0, stream>>>(srvOut, decOut, tem);

  // ---- graph normalization (shared by both ST blocks) ----
  hipMemsetAsync(deg, 0, MS * sizeof(float), stream);
  deg_kernel<<<eg(NE, 256), 256, 0, stream>>>(edge, deg);
  dis_kernel<<<1, 128, 0, stream>>>(deg);
  w_kernel<<<eg(NE, 256), 256, 0, stream>>>(edge, edge + NE, deg, wE);

  // _tconv: 3 gated convs, each = 3 accumulated VALID taps via strided WMMA GEMM
  auto tconv = [&](const float* X, int H, int Ci, int Co, const float* const* W,
                   const float* const* Bv, float* P, float* Q, float* R, float* O, float scale) {
    const int Ho = H - 2, M = TT * Ho * MS;
    float* outs[3] = {P, Q, R};
    for (int j = 0; j < 3; ++j)
      for (int r = 0; r < 3; ++r)
        gemm3(X + (long)r * MS * Ci, nullptr, TT, Ho, MS,
              (long)H * MS * Ci, (long)MS * Ci, Ci,
              mkT(W[j] + (long)r * Ci * Co, Ci, Co), r == 0 ? Bv[j] : nullptr, nullptr,
              outs[j], Co, M, Co, Ci, r == 0 ? 0 : 2);
    glu_act<<<eg((long)M * Co, 256), 256, 0, stream>>>(P, Q, R, O, (long)M * Co, scale);
  };
  // _cheb (K=3): node-major transpose, 3 WMMA term-GEMMs + atomic scatter props
  auto cheb = [&](const float* X, int H, const float* cW, const float* cb,
                  float* Z0, float* Z1, float* Pp, float* Z2, float* Out, float* Xo) {
    const int S = TT * H * 128, M = MS * TT * H;
    const long tot = (long)M * 128;
    to_nodes<<<eg(tot, 256), 256, 0, stream>>>(X, Z0, TT, H, MS, 128);
    gemm(Z0, 128, mkT(cW, 128, 128), cb, nullptr, Out, 128, M, 128, 128, 0);
    hipMemsetAsync(Z1, 0, tot * sizeof(float), stream);
    cheb_prop<<<eg((long)NE * S, 256), 256, 0, stream>>>(edge, edge + NE, wE, Z0, Z1, S);
    gemm(Z1, 128, mkT(cW + 128 * 128, 128, 128), nullptr, nullptr, Out, 128, M, 128, 128, 2);
    hipMemsetAsync(Pp, 0, tot * sizeof(float), stream);
    cheb_prop<<<eg((long)NE * S, 256), 256, 0, stream>>>(edge, edge + NE, wE, Z1, Pp, S);
    z2_combine<<<eg(tot, 256), 256, 0, stream>>>(Pp, Z0, Z2, tot);
    gemm(Z2, 128, mkT(cW + 2 * 128 * 128, 128, 128), nullptr, nullptr, Out, 128, M, 128, 128, 2);
    from_nodes_relu<<<eg(tot, 256), 256, 0, stream>>>(Out, Xo, TT, H, MS, 128);
  };

  const float invs = 0.99999500003749969f;  // 1/sqrt(1+1e-5)
  // ST block 1: (T,9,m,128) -> t1 -> (T,7,m,128) -> cheb -> t2 -> (T,5,m,64)
  tconv(tem, 9, 128, 128, st1_t1W, st1_t1b, cP, cQ, cR, cX, 1.f);
  cheb(cX, 7, st1_cW, st1_cb, Za, Zb, Zp, Zc, OutC, Xr);
  tconv(Xr, 7, 128, 64, st1_t2W, st1_t2b, dP, dQ, dR, Y1, invs);
  // ST block 2: (T,5,m,64) -> t1 -> (T,3,m,128) -> cheb -> t2 -> (T,1,m,64)
  tconv(Y1, 5, 64, 128, st2_t1W, st2_t1b, eP, eQ, eR, eX, 1.f);
  cheb(eX, 3, st2_cW, st2_cb, Za2, Zb2, Zp2, Zc2, OutC2, Xr2);
  tconv(Xr2, 3, 128, 64, st2_t2W, st2_t2b, fP, fQ, fR, tem2, invs);

  // ---- final gather + QoS MLP (192->32->16->8->2) ----
  build_final<<<eg((long)BB * 192, 256), 256, 0, stream>>>(usrOut, tem2, svcEmb, info, inv, xF);
  gemm(xF, 192, mkT(qW0, 192, 32), qb0, nullptr, m1, 32, BB, 32, 192, 1);
  gemm(m1, 32, mkT(qW1, 32, 16), qb1, nullptr, m2, 16, BB, 16, 32, 1);
  gemm(m2, 16, mkT(qW2, 16, 8), qb2, nullptr, m3, 8, BB, 8, 16, 1);
  gemm(m3, 8, mkT(qW3, 8, 2), qb3, nullptr, (float*)d_out, 2, BB, 2, 8, 0);
}
